// Block_23244363005986
// MI455X (gfx1250) — compile-verified, matrix operand order checked by reading, not run on hardware
//
#include <hip/hip_runtime.h>
#include <hip/hip_bf16.h>

typedef __attribute__((ext_vector_type(16))) __bf16 v16bf;
typedef __attribute__((ext_vector_type(8)))  __bf16 v8bf;
typedef __attribute__((ext_vector_type(4)))  __bf16 v4bf;
typedef __attribute__((ext_vector_type(8)))  float  v8f;

constexpr int Bv   = 2;
constexpr int Tv   = 2048;
constexpr int Cv   = 1024;
constexpr int NHv  = 16;
constexpr int HDv  = 64;
constexpr int Ev   = 8;
constexpr int HIDv = 4096;
constexpr int NTOK = Bv * Tv;          // 4096

#define GEMM_ASYNC_STAGE 1

// ---------------------------------------------------------------- helpers
static __device__ __forceinline__ v8f wmma_bf16(v16bf a, v16bf b, v8f c) {
  return __builtin_amdgcn_wmma_f32_16x16x32_bf16(false, a, false, b, (short)0, c,
                                                 false, false);
}

static __device__ __forceinline__ v8f zero8() {
  return (v8f){0.f, 0.f, 0.f, 0.f, 0.f, 0.f, 0.f, 0.f};
}

// Async global->LDS 16B copy (ASYNCcnt-tracked, no VGPR data round-trip).
static __device__ __forceinline__ void async_b128(const __bf16* gptr,
                                                  __bf16* lptr) {
  asm volatile("global_load_async_to_lds_b128 %0, %1, off"
               :
               : "v"((unsigned)(unsigned long long)lptr),
                 "v"((unsigned long long)gptr)
               : "memory");
}
static __device__ __forceinline__ void wait_async0() {
  asm volatile("s_wait_asynccnt 0" ::: "memory");
}

// Fragment from a K-contiguous row (LDS or global). Element e holds
// k = hi8 + e (e<8) and k = hi8 + 16 + (e-8) (e>=8)  -> two aligned 16B loads.
static __device__ __forceinline__ v16bf frag_row(const __bf16* rowp, int hi8) {
  v8bf lo = *(const v8bf*)(rowp + hi8);
  v8bf hi = *(const v8bf*)(rowp + hi8 + 16);
  return __builtin_shufflevector(lo, hi, 0, 1, 2, 3, 4, 5, 6, 7, 8, 9, 10, 11,
                                 12, 13, 14, 15);
}

// ---------------------------------------------------------------- fp32 -> bf16 (+T)
// dst[N,K] = (bf16) src[K,N]^T  ; 32x32 LDS tiles.
__global__ void cvt_t_kernel(const float* __restrict__ src,
                             __bf16* __restrict__ dst, int K, int N) {
  __shared__ __bf16 L[32][33];
  const int tilesN = N >> 5;
  const int kt = ((int)blockIdx.x / tilesN) << 5;
  const int nt = ((int)blockIdx.x % tilesN) << 5;
  const int r  = threadIdx.x >> 3;
  const int c4 = (threadIdx.x & 7) << 2;
#pragma unroll
  for (int j = 0; j < 4; ++j)
    L[r][c4 + j] = (__bf16)src[(size_t)(kt + r) * N + nt + c4 + j];
  __syncthreads();
#pragma unroll
  for (int j = 0; j < 4; ++j)
    dst[(size_t)(nt + r) * K + kt + c4 + j] = L[c4 + j][r];
}

// ---------------------------------------------------------------- RMSNorm
__global__ void rmsnorm_kernel(const float* __restrict__ x,
                               const float* __restrict__ w,
                               __bf16* __restrict__ out) {
  __shared__ float red[8];
  const float* xr = x + (size_t)blockIdx.x * Cv;
  float ss = 0.f;
  for (int c = threadIdx.x; c < Cv; c += 256) { float v = xr[c]; ss += v * v; }
#pragma unroll
  for (int s = 16; s > 0; s >>= 1) ss += __shfl_xor(ss, s, 32);
  if ((threadIdx.x & 31) == 0) red[threadIdx.x >> 5] = ss;
  __syncthreads();
  float tot = 0.f;
#pragma unroll
  for (int i = 0; i < 8; ++i) tot += red[i];
  const float inv = rsqrtf(tot / (float)Cv + 1e-8f);
  __bf16* o = out + (size_t)blockIdx.x * Cv;
  for (int c = threadIdx.x; c < Cv; c += 256) o[c] = (__bf16)(xr[c] * inv * w[c]);
}

// ---------------------------------------------------------------- GEMM (bf16 WMMA)
// out[M,N] = A[M,K] @ WT[N,K]^T + bias (+resid). 256 thr = 8 waves.
// Block tile 64x128; wave tile 32x32 (2x2 WMMA, register-blocked).
__global__ void gemm_kernel(const __bf16* __restrict__ A,
                            const __bf16* __restrict__ WT,  // [N,K]
                            const float* __restrict__ bias,
                            const float* __restrict__ resid,
                            __bf16* __restrict__ outb,
                            float* __restrict__ outf,
                            int M, int N, int K) {
  __shared__ __bf16 As[64][40];
  __shared__ __bf16 Bs[128][40];
  const int tid  = threadIdx.x;
  const int lane = tid & 31;
  const int wave = tid >> 5;
  const int tilesN = N >> 7;
  const int bm = ((int)blockIdx.x / tilesN) << 6;
  const int bn = ((int)blockIdx.x % tilesN) << 7;
  const int wm = (wave & 1) << 5;
  const int wn = (wave >> 1) << 5;
  const int ml  = lane & 15;
  const int hi8 = (lane >> 4) << 3;
  v8f acc[2][2];
#pragma unroll
  for (int mi = 0; mi < 2; ++mi)
#pragma unroll
    for (int ni = 0; ni < 2; ++ni) acc[mi][ni] = zero8();

  const int ra = tid >> 2, ca = (tid & 3) << 3;   // A: 64x32, 8/thread
  const int rb = tid >> 1, cb = (tid & 1) << 4;   // B: 128x32, 16/thread

  for (int kk = 0; kk < K; kk += 32) {
    const __bf16* sa = A + (size_t)(bm + ra) * K + kk + ca;
    const __bf16* sb = WT + (size_t)(bn + rb) * K + kk + cb;
#if GEMM_ASYNC_STAGE
    async_b128(sa, &As[ra][ca]);
    async_b128(sb, &Bs[rb][cb]);
    async_b128(sb + 8, &Bs[rb][cb + 8]);
#else
    *(v8bf*)&As[ra][ca]     = *(const v8bf*)sa;
    *(v8bf*)&Bs[rb][cb]     = *(const v8bf*)sb;
    *(v8bf*)&Bs[rb][cb + 8] = *(const v8bf*)(sb + 8);
#endif
    if (kk + 32 < K) __builtin_prefetch(sb + 32, 0, 3);
#if GEMM_ASYNC_STAGE
    wait_async0();
#endif
    __syncthreads();
    v16bf af[2], bf[2];
#pragma unroll
    for (int mi = 0; mi < 2; ++mi)
      af[mi] = frag_row(&As[wm + mi * 16 + ml][0], hi8);
#pragma unroll
    for (int ni = 0; ni < 2; ++ni)
      bf[ni] = frag_row(&Bs[wn + ni * 16 + ml][0], hi8);
#pragma unroll
    for (int mi = 0; mi < 2; ++mi)
#pragma unroll
      for (int ni = 0; ni < 2; ++ni)
        acc[mi][ni] = wmma_bf16(af[mi], bf[ni], acc[mi][ni]);
    __syncthreads();
  }

#pragma unroll
  for (int mi = 0; mi < 2; ++mi)
#pragma unroll
    for (int ni = 0; ni < 2; ++ni) {
      const int n  = bn + wn + ni * 16 + ml;
      const int m0 = bm + wm + mi * 16 + hi8;
      const float bvv = bias ? bias[n] : 0.f;
#pragma unroll
      for (int r = 0; r < 8; ++r) {
        const int m = m0 + r;
        float v = acc[mi][ni][r] + bvv;
        if (resid) v += resid[(size_t)m * N + n];
        if (outf)  outf[(size_t)m * N + n] = v;
        if (outb)  outb[(size_t)m * N + n] = (__bf16)v;
      }
    }
}

// ---------------------------------------------------------------- V transpose
// vT[(b*NH+h)*HD + d][t] = V[b,t,h,d]  (V = kv[:, C + h*64 + d])
__global__ void transpose_v_kernel(const __bf16* __restrict__ kv,
                                   __bf16* __restrict__ vT) {
  __shared__ __bf16 L[64][65];
  const int tiles = Tv >> 6;                  // 32
  const int bh = (int)blockIdx.x / tiles;     // 0..31
  const int kt = ((int)blockIdx.x % tiles) << 6;
  const int h = bh & 15, b = bh >> 4;
#pragma unroll
  for (int i = 0; i < 16; ++i) {
    const int flat = threadIdx.x + (i << 8);
    const int r = flat >> 6, c = flat & 63;   // r = key-in-tile, c = d
    L[r][c] = kv[((size_t)(b * Tv) + kt + r) * (2 * Cv) + Cv + h * HDv + c];
  }
  __syncthreads();
#pragma unroll
  for (int i = 0; i < 16; ++i) {
    const int flat = threadIdx.x + (i << 8);
    const int r = flat >> 6, c = flat & 63;   // r = d, c = key-in-tile
    vT[((size_t)bh * HDv + r) * Tv + kt + c] = L[c][r];
  }
}

// ---------------------------------------------------------------- attention
// One wave per 16-query tile; flash-style over 32-key chunks with ALiBi+causal.
__global__ void attn_kernel(const __bf16* __restrict__ qmat,   // [NTOK, C]
                            const __bf16* __restrict__ kvmat,  // [NTOK, 2C]
                            const __bf16* __restrict__ vT,     // [B*NH*HD, T]
                            __bf16* __restrict__ y) {          // [NTOK, C]
  __shared__ __bf16 Ps[8][16][32];
  const int lane = threadIdx.x & 31;
  const int wave = threadIdx.x >> 5;
  const int task = blockIdx.x * 8 + wave;   // B*NH*(T/16) = 4096 exactly
  const int qt = task & 127;
  const int bh = task >> 7;
  const int h  = bh & 15;
  const int b  = bh >> 4;
  const int qbase = qt << 4;
  const float slope = (float)(h + 1) / (float)NHv;
  const int ml  = lane & 15;
  const int hi8 = (lane >> 4) << 3;

  // Q fragments (16x64 -> two K-steps of 32)
  const __bf16* qr = qmat + ((size_t)(b * Tv) + qbase + ml) * Cv + h * HDv;
  v16bf qf0 = frag_row(qr, hi8);
  v16bf qf1 = frag_row(qr + 32, hi8);

  v8f oacc[4];
#pragma unroll
  for (int d = 0; d < 4; ++d) oacc[d] = zero8();
  float mrun[8], lrun[8];
#pragma unroll
  for (int r = 0; r < 8; ++r) { mrun[r] = -1e30f; lrun[r] = 0.f; }

  for (int kb = 0; kb < qbase + 16; kb += 32) {
    // ---- S = Q @ K^T (two 16x16 key tiles)
    v8f s01[2];
#pragma unroll
    for (int j = 0; j < 2; ++j) {
      const __bf16* kr =
          kvmat + ((size_t)(b * Tv) + kb + j * 16 + ml) * (2 * Cv) + h * HDv;
      v16bf k0 = frag_row(kr, hi8);
      v16bf k1 = frag_row(kr + 32, hi8);
      v8f s = zero8();
      s = wmma_bf16(qf0, k0, s);
      s = wmma_bf16(qf1, k1, s);
      s01[j] = s;
    }

    // ---- masked ALiBi softmax, streaming
#pragma unroll
    for (int r = 0; r < 8; ++r) {
      const int qrow = qbase + r + hi8;
      const int key0 = kb + ml, key1 = kb + 16 + ml;
      float a0 = (key0 <= qrow)
                     ? s01[0][r] * 0.125f + slope * (float)(key0 - qrow)
                     : -1e30f;
      float a1 = (key1 <= qrow)
                     ? s01[1][r] * 0.125f + slope * (float)(key1 - qrow)
                     : -1e30f;
      float rm = fmaxf(a0, a1);
#pragma unroll
      for (int s = 8; s > 0; s >>= 1) rm = fmaxf(rm, __shfl_xor(rm, s, 32));
      const float mn = fmaxf(mrun[r], rm);
      const float sc = __expf(mrun[r] - mn);
      const float e0 = __expf(a0 - mn);
      const float e1 = __expf(a1 - mn);
      float rs = e0 + e1;
#pragma unroll
      for (int s = 8; s > 0; s >>= 1) rs += __shfl_xor(rs, s, 32);
      lrun[r] = lrun[r] * sc + rs;
      mrun[r] = mn;
#pragma unroll
      for (int d = 0; d < 4; ++d) oacc[d][r] *= sc;
      Ps[wave][r + hi8][ml]      = (__bf16)e0;
      Ps[wave][r + hi8][16 + ml] = (__bf16)e1;
    }
    __builtin_amdgcn_wave_barrier();

    // ---- P (D-layout -> A-layout via LDS), then O += P @ V
    v16bf pf = frag_row(&Ps[wave][ml][0], hi8);
#pragma unroll
    for (int d4 = 0; d4 < 4; ++d4) {
      const __bf16* vr = vT + ((size_t)bh * HDv + d4 * 16 + ml) * Tv + kb;
      v16bf vf = frag_row(vr, hi8);
      oacc[d4] = wmma_bf16(pf, vf, oacc[d4]);
    }
    __builtin_amdgcn_wave_barrier();
  }

  // ---- write y (bf16), normalized by l
#pragma unroll
  for (int d4 = 0; d4 < 4; ++d4)
#pragma unroll
    for (int r = 0; r < 8; ++r) {
      const int m = qbase + r + hi8;
      const float v = oacc[d4][r] / lrun[r];
      y[((size_t)(b * Tv) + m) * Cv + h * HDv + d4 * 16 + ml] = (__bf16)v;
    }
}

// ---------------------------------------------------------------- router (top-2)
__global__ void zero_counts_kernel(int* counts) {
  if (threadIdx.x < Ev) counts[threadIdx.x] = 0;
}

__global__ void router_kernel(const __bf16* __restrict__ h2,
                              const float* __restrict__ rw,  // [C, E]
                              int* __restrict__ counts,
                              int* __restrict__ idx,   // [E, NTOK]
                              float* __restrict__ wgt) // [E, NTOK]
{
  const int lane = threadIdx.x & 31;
  const int wave = threadIdx.x >> 5;
  const int t = blockIdx.x * 8 + wave;
  if (t >= NTOK) return;
  float lg[Ev];
#pragma unroll
  for (int e = 0; e < Ev; ++e) lg[e] = 0.f;
  const __bf16* hp = h2 + (size_t)t * Cv;
  for (int c = lane; c < Cv; c += 32) {
    const float hv = (float)hp[c];
    const float* rp = rw + (size_t)c * Ev;
#pragma unroll
    for (int e = 0; e < Ev; ++e) lg[e] += hv * rp[e];
  }
#pragma unroll
  for (int e = 0; e < Ev; ++e)
#pragma unroll
    for (int s = 16; s > 0; s >>= 1) lg[e] += __shfl_xor(lg[e], s, 32);
  if (lane == 0) {
    int i1 = 0;
    for (int e = 1; e < Ev; ++e) if (lg[e] > lg[i1]) i1 = e;
    int i2 = -1;
    for (int e = 0; e < Ev; ++e)
      if (e != i1 && (i2 < 0 || lg[e] > lg[i2])) i2 = e;
    const float e2 = __expf(lg[i2] - lg[i1]);
    const float w1 = 1.f / (1.f + e2);
    const float w2 = e2 / (1.f + e2);
    int p1 = atomicAdd(&counts[i1], 1);
    idx[i1 * NTOK + p1] = t; wgt[i1 * NTOK + p1] = w1;
    int p2 = atomicAdd(&counts[i2], 1);
    idx[i2 * NTOK + p2] = t; wgt[i2 * NTOK + p2] = w2;
  }
}

// ---------------------------------------------------------------- MoE gate||up
// gi[r, HID] = silu(h2[tok_r] @ Wg) * (h2[tok_r] @ Wu), gathered rows.
// Block tile 64x128; wave tile 32x32 for BOTH gate and up (8 WMMA / K-step).
__global__ void moe_gateup_kernel(const __bf16* __restrict__ A,    // [NTOK, C]
                                  const __bf16* __restrict__ WgT,  // [HID, C]
                                  const __bf16* __restrict__ WuT,  // [HID, C]
                                  const int* __restrict__ cnt_p,
                                  const int* __restrict__ gidx,
                                  __bf16* __restrict__ gi)         // [NTOK, HID]
{
  const int cnt = *cnt_p;
  const int tilesN = HIDv >> 7;
  const int bm = ((int)blockIdx.x / tilesN) << 6;
  if (bm >= cnt) return;
  const int bn = ((int)blockIdx.x % tilesN) << 7;

  __shared__ __bf16 As[64][40];
  __shared__ __bf16 Bg[128][40];
  __shared__ __bf16 Bu[128][40];
  const int tid  = threadIdx.x;
  const int lane = tid & 31;
  const int wave = tid >> 5;
  const int wm = (wave & 1) << 5;
  const int wn = (wave >> 1) << 5;
  const int ml  = lane & 15;
  const int hi8 = (lane >> 4) << 3;
  v8f accg[2][2], accu[2][2];
#pragma unroll
  for (int mi = 0; mi < 2; ++mi)
#pragma unroll
    for (int ni = 0; ni < 2; ++ni) { accg[mi][ni] = zero8(); accu[mi][ni] = zero8(); }

  const int ra = tid >> 2, ca = (tid & 3) << 3;
  const int rb = tid >> 1, cb = (tid & 1) << 4;
  const int rowa = bm + ra;
  const int tok = gidx[rowa < cnt ? rowa : cnt - 1];

  for (int kk = 0; kk < Cv; kk += 32) {
    const __bf16* sa = A + (size_t)tok * Cv + kk + ca;
    const __bf16* sg = WgT + (size_t)(bn + rb) * Cv + kk + cb;
    const __bf16* su = WuT + (size_t)(bn + rb) * Cv + kk + cb;
#if GEMM_ASYNC_STAGE
    async_b128(sa, &As[ra][ca]);
    async_b128(sg, &Bg[rb][cb]);
    async_b128(sg + 8, &Bg[rb][cb + 8]);
    async_b128(su, &Bu[rb][cb]);
    async_b128(su + 8, &Bu[rb][cb + 8]);
#else
    *(v8bf*)&As[ra][ca]     = *(const v8bf*)sa;
    *(v8bf*)&Bg[rb][cb]     = *(const v8bf*)sg;
    *(v8bf*)&Bg[rb][cb + 8] = *(const v8bf*)(sg + 8);
    *(v8bf*)&Bu[rb][cb]     = *(const v8bf*)su;
    *(v8bf*)&Bu[rb][cb + 8] = *(const v8bf*)(su + 8);
#endif
    if (kk + 32 < Cv) { __builtin_prefetch(sg + 32, 0, 3); __builtin_prefetch(su + 32, 0, 3); }
#if GEMM_ASYNC_STAGE
    wait_async0();
#endif
    __syncthreads();
    v16bf af[2], bg[2], bu[2];
#pragma unroll
    for (int mi = 0; mi < 2; ++mi)
      af[mi] = frag_row(&As[wm + mi * 16 + ml][0], hi8);
#pragma unroll
    for (int ni = 0; ni < 2; ++ni) {
      bg[ni] = frag_row(&Bg[wn + ni * 16 + ml][0], hi8);
      bu[ni] = frag_row(&Bu[wn + ni * 16 + ml][0], hi8);
    }
#pragma unroll
    for (int mi = 0; mi < 2; ++mi)
#pragma unroll
      for (int ni = 0; ni < 2; ++ni) {
        accg[mi][ni] = wmma_bf16(af[mi], bg[ni], accg[mi][ni]);
        accu[mi][ni] = wmma_bf16(af[mi], bu[ni], accu[mi][ni]);
      }
    __syncthreads();
  }

#pragma unroll
  for (int mi = 0; mi < 2; ++mi)
#pragma unroll
    for (int ni = 0; ni < 2; ++ni) {
      const int n  = bn + wn + ni * 16 + ml;
      const int m0 = bm + wm + mi * 16 + hi8;
#pragma unroll
      for (int r = 0; r < 8; ++r) {
        const int m = m0 + r;
        const float g = accg[mi][ni][r];
        const float u = accu[mi][ni][r];
        const float silu = g / (1.f + __expf(-g));
        gi[(size_t)m * HIDv + n] = (__bf16)(silu * u);
      }
    }
}

// ---------------------------------------------------------------- MoE down + scatter
__global__ void moe_down_kernel(const __bf16* __restrict__ gi,   // [NTOK, HID]
                                const __bf16* __restrict__ WdT,  // [C, HID]
                                const int* __restrict__ cnt_p,
                                const int* __restrict__ gidx,
                                const float* __restrict__ gwgt,
                                float* __restrict__ out)         // [NTOK, C]
{
  const int cnt = *cnt_p;
  const int tilesN = Cv >> 7;
  const int bm = ((int)blockIdx.x / tilesN) << 6;
  if (bm >= cnt) return;
  const int bn = ((int)blockIdx.x % tilesN) << 7;

  __shared__ __bf16 As[64][40];
  __shared__ __bf16 Bs[128][40];
  const int tid  = threadIdx.x;
  const int lane = tid & 31;
  const int wave = tid >> 5;
  const int wm = (wave & 1) << 5;
  const int wn = (wave >> 1) << 5;
  const int ml  = lane & 15;
  const int hi8 = (lane >> 4) << 3;
  v8f acc[2][2];
#pragma unroll
  for (int mi = 0; mi < 2; ++mi)
#pragma unroll
    for (int ni = 0; ni < 2; ++ni) acc[mi][ni] = zero8();

  const int ra = tid >> 2, ca = (tid & 3) << 3;
  const int rb = tid >> 1, cb = (tid & 1) << 4;

  for (int kk = 0; kk < HIDv; kk += 32) {
    const __bf16* sa = gi + (size_t)(bm + ra) * HIDv + kk + ca;
    const __bf16* sb = WdT + (size_t)(bn + rb) * HIDv + kk + cb;
#if GEMM_ASYNC_STAGE
    async_b128(sa, &As[ra][ca]);
    async_b128(sb, &Bs[rb][cb]);
    async_b128(sb + 8, &Bs[rb][cb + 8]);
#else
    *(v8bf*)&As[ra][ca]     = *(const v8bf*)sa;
    *(v8bf*)&Bs[rb][cb]     = *(const v8bf*)sb;
    *(v8bf*)&Bs[rb][cb + 8] = *(const v8bf*)(sb + 8);
#endif
    if (kk + 32 < HIDv) __builtin_prefetch(sb + 32, 0, 3);
#if GEMM_ASYNC_STAGE
    wait_async0();
#endif
    __syncthreads();
    v16bf af[2], bf[2];
#pragma unroll
    for (int mi = 0; mi < 2; ++mi)
      af[mi] = frag_row(&As[wm + mi * 16 + ml][0], hi8);
#pragma unroll
    for (int ni = 0; ni < 2; ++ni)
      bf[ni] = frag_row(&Bs[wn + ni * 16 + ml][0], hi8);
#pragma unroll
    for (int mi = 0; mi < 2; ++mi)
#pragma unroll
      for (int ni = 0; ni < 2; ++ni)
        acc[mi][ni] = wmma_bf16(af[mi], bf[ni], acc[mi][ni]);
    __syncthreads();
  }

#pragma unroll
  for (int mi = 0; mi < 2; ++mi)
#pragma unroll
    for (int ni = 0; ni < 2; ++ni) {
      const int n  = bn + wn + ni * 16 + ml;
      const int m0 = bm + wm + mi * 16 + hi8;
#pragma unroll
      for (int r = 0; r < 8; ++r) {
        const int m = m0 + r;
        if (m < cnt)
          atomicAdd(&out[(size_t)gidx[m] * Cv + n], gwgt[m] * acc[mi][ni][r]);
      }
    }
}

// ---------------------------------------------------------------- launch
extern "C" void kernel_launch(void* const* d_in, const int* in_sizes, int n_in,
                              void* d_out, int out_size, void* d_ws,
                              size_t ws_size, hipStream_t stream) {
  const float* x           = (const float*)d_in[0];
  const float* attn_norm_w = (const float*)d_in[1];
  const float* ffn_norm_w  = (const float*)d_in[2];
  const float* qw          = (const float*)d_in[3];
  const float* qb          = (const float*)d_in[4];
  const float* kvw         = (const float*)d_in[5];
  const float* kvb         = (const float*)d_in[6];
  const float* ow          = (const float*)d_in[7];
  const float* ob          = (const float*)d_in[8];
  const float* router_w    = (const float*)d_in[9];
  const float* gate_w      = (const float*)d_in[10];
  const float* up_w        = (const float*)d_in[11];
  const float* down_w      = (const float*)d_in[12];
  float* out = (float*)d_out;

  char* p = (char*)d_ws;
  auto carve = [&](size_t bytes) {
    char* r = p;
    p += (bytes + 255) & ~(size_t)255;
    return r;
  };
  __bf16* h_bf    = (__bf16*)carve((size_t)NTOK * Cv * 2);
  __bf16* q_bf    = (__bf16*)carve((size_t)NTOK * Cv * 2);
  __bf16* kv_bf   = (__bf16*)carve((size_t)NTOK * 2 * Cv * 2);
  __bf16* vT_bf   = (__bf16*)carve((size_t)Bv * NHv * HDv * Tv * 2);
  __bf16* y_bf    = (__bf16*)carve((size_t)NTOK * Cv * 2);
  __bf16* h2_bf   = (__bf16*)carve((size_t)NTOK * Cv * 2);
  __bf16* qwT_bf  = (__bf16*)carve((size_t)Cv * Cv * 2);
  __bf16* kvwT_bf = (__bf16*)carve((size_t)Cv * 2 * Cv * 2);
  __bf16* owT_bf  = (__bf16*)carve((size_t)Cv * Cv * 2);
  __bf16* wgT_bf  = (__bf16*)carve((size_t)Cv * HIDv * 2);
  __bf16* wuT_bf  = (__bf16*)carve((size_t)Cv * HIDv * 2);
  __bf16* wdT_bf  = (__bf16*)carve((size_t)HIDv * Cv * 2);
  __bf16* gi_bf   = (__bf16*)carve((size_t)NTOK * HIDv * 2);
  int*    counts  = (int*)carve(Ev * sizeof(int));
  int*    eidx    = (int*)carve((size_t)Ev * NTOK * sizeof(int));
  float*  ewgt    = (float*)carve((size_t)Ev * NTOK * sizeof(float));

  const dim3 blk(256);

  // weight convert+transpose (attention path): dst[N,K] from src[K,N]
  cvt_t_kernel<<<(Cv / 32) * (Cv / 32), blk, 0, stream>>>(qw, qwT_bf, Cv, Cv);
  cvt_t_kernel<<<(Cv / 32) * (2 * Cv / 32), blk, 0, stream>>>(kvw, kvwT_bf, Cv,
                                                              2 * Cv);
  cvt_t_kernel<<<(Cv / 32) * (Cv / 32), blk, 0, stream>>>(ow, owT_bf, Cv, Cv);

  // h = rmsnorm(x)
  rmsnorm_kernel<<<NTOK, blk, 0, stream>>>(x, attn_norm_w, h_bf);

  // q = h@qw + qb ; kv = h@kvw + kvb
  gemm_kernel<<<(NTOK / 64) * (Cv / 128), blk, 0, stream>>>(
      h_bf, qwT_bf, qb, nullptr, q_bf, nullptr, NTOK, Cv, Cv);
  gemm_kernel<<<(NTOK / 64) * (2 * Cv / 128), blk, 0, stream>>>(
      h_bf, kvwT_bf, kvb, nullptr, kv_bf, nullptr, NTOK, 2 * Cv, Cv);

  // V -> [B*NH*HD, T] for K-contiguous PV fragments
  transpose_v_kernel<<<Bv * NHv * (Tv / 64), blk, 0, stream>>>(kv_bf, vT_bf);

  // attention
  attn_kernel<<<(Bv * NHv * (Tv / 16)) / 8, blk, 0, stream>>>(q_bf, kv_bf,
                                                              vT_bf, y_bf);

  // x2 = x + y@ow + ob  -> d_out (f32)
  gemm_kernel<<<(NTOK / 64) * (Cv / 128), blk, 0, stream>>>(
      y_bf, owT_bf, ob, x, nullptr, out, NTOK, Cv, Cv);

  // h2 = rmsnorm(x2)
  rmsnorm_kernel<<<NTOK, blk, 0, stream>>>(out, ffn_norm_w, h2_bf);

  // router: top-2 token lists per expert
  zero_counts_kernel<<<1, 64, 0, stream>>>(counts);
  router_kernel<<<NTOK / 8, blk, 0, stream>>>(h2_bf, router_w, counts, eidx,
                                              ewgt);

  // MoE: per-expert gathered GEMMs, scatter-add into d_out
  for (int e = 0; e < Ev; ++e) {
    cvt_t_kernel<<<(Cv / 32) * (HIDv / 32), blk, 0, stream>>>(
        gate_w + (size_t)e * Cv * HIDv, wgT_bf, Cv, HIDv);
    cvt_t_kernel<<<(Cv / 32) * (HIDv / 32), blk, 0, stream>>>(
        up_w + (size_t)e * Cv * HIDv, wuT_bf, Cv, HIDv);
    cvt_t_kernel<<<(HIDv / 32) * (Cv / 32), blk, 0, stream>>>(
        down_w + (size_t)e * HIDv * Cv, wdT_bf, HIDv, Cv);
    moe_gateup_kernel<<<(NTOK / 64) * (HIDv / 128), blk, 0, stream>>>(
        h2_bf, wgT_bf, wuT_bf, counts + e, eidx + (size_t)e * NTOK, gi_bf);
    moe_down_kernel<<<(NTOK / 64) * (Cv / 128), blk, 0, stream>>>(
        gi_bf, wdT_bf, counts + e, eidx + (size_t)e * NTOK,
        ewgt + (size_t)e * NTOK, out);
  }
}
